// DCN_5214090298004
// MI455X (gfx1250) — compile-verified, matrix operand order checked by reading, not run on hardware
//
#include <hip/hip_runtime.h>
#include <hip/hip_bf16.h>

// ---------------------------------------------------------------------------
// DCNv2 forward for MI455X (gfx1250, wave32, WMMA + TDM).
//   B=4, C=256, H=W=64, O=256, K=3, stride=1, pad=1, dil=1, DG=1
// Pipeline:
//   k1: offset/mask conv  -> om[B,27,4096]        (f32, sigmoid on mask chans)
//   k2: bilinear im2col   -> colsT[B,4096,2304]   (f16, K reordered kk-major)
//   k3: weight convert    -> Wf16[256,2304]       (f16, same K reordering)
//   k4: WMMA GEMM + bias  -> out[B,256,4096]      (f32, TDM double-buffered)
// ---------------------------------------------------------------------------

typedef __attribute__((ext_vector_type(16))) _Float16 v16h;
typedef __attribute__((ext_vector_type(8)))  _Float16 v8h;
typedef __attribute__((ext_vector_type(8)))  float    v8f;
typedef __attribute__((ext_vector_type(4)))  unsigned int u32x4;
typedef __attribute__((ext_vector_type(8)))  int      i32x8;
typedef __attribute__((ext_vector_type(4)))  int      i32x4;

#define BB 4
#define CC 256
#define HH 64
#define WW 64
#define OO 256
#define NN 4096              // Ho*Wo
#define KK9 9
#define KDIM 2304            // C*9
#define HW 4096              // H*W

union FragAB { v16h v; v8h h[2]; };

#if defined(__gfx1250__) && __has_builtin(__builtin_amdgcn_tensor_load_to_lds) && \
    __has_builtin(__builtin_amdgcn_s_wait_tensorcnt)
#define HAVE_TDM 1
#else
#define HAVE_TDM 0
#endif

// --------------------------- kernel 1: om conv -----------------------------
__global__ void dcn_om_conv(const float* __restrict__ x,
                            const float* __restrict__ w_om,
                            const float* __restrict__ b_om,
                            float* __restrict__ om) {
    int idx = blockIdx.x * blockDim.x + threadIdx.x;
    if (idx >= BB * 27 * NN) return;
    int n  = idx & (NN - 1);
    int ch = (idx >> 12) % 27;
    int b  = idx / (27 * NN);
    int ho = n >> 6, wo = n & 63;

    float acc = b_om[ch];
    const float* wch = w_om + (size_t)ch * KDIM;       // [c][kh][kw]
    const float* xb  = x + (size_t)b * CC * HW;

    int kh0 = (ho == 0) ? 1 : 0, kh1 = (ho == HH - 1) ? 2 : 3;
    int kw0 = (wo == 0) ? 1 : 0, kw1 = (wo == WW - 1) ? 2 : 3;

    for (int c = 0; c < CC; ++c) {
        const float* xc = xb + c * HW;
        const float* wc = wch + c * 9;
        for (int kh = kh0; kh < kh1; ++kh) {
            const float* xr = xc + (ho - 1 + kh) * WW + (wo - 1);
            const float* wr = wc + kh * 3;
            for (int kw = kw0; kw < kw1; ++kw)
                acc += wr[kw] * xr[kw];
        }
    }
    if (ch >= 18) acc = 1.0f / (1.0f + __expf(-acc));   // sigmoid for mask
    om[((size_t)b * 27 + ch) * NN + n] = acc;
}

// --------------------------- kernel 2: im2col ------------------------------
// colsT[b][n][kk*256 + c]  (B^T of the GEMM B matrix, K contiguous per row)
__global__ void dcn_im2col(const float* __restrict__ x,
                           const float* __restrict__ om,
                           _Float16* __restrict__ colsT) {
    int idx = blockIdx.x * blockDim.x + threadIdx.x;
    if (idx >= BB * KK9 * NN) return;
    int n  = idx & (NN - 1);
    int kk = (idx >> 12) % KK9;
    int b  = idx / (KK9 * NN);

    const size_t omb = (size_t)b * 27 * NN;
    float dy = om[omb + (size_t)kk * NN + n];
    float dx = om[omb + (size_t)(9 + kk) * NN + n];
    float mk = om[omb + (size_t)(18 + kk) * NN + n];    // sigmoid already done

    int ho = n >> 6, wo = n & 63;
    float py = (float)(ho - 1 + kk / 3) + dy;
    float px = (float)(wo - 1 + kk % 3) + dx;
    float y0f = floorf(py), x0f = floorf(px);
    float ly = py - y0f, lx = px - x0f;
    int y0 = (int)y0f, x0 = (int)x0f;
    int y1 = y0 + 1,   x1 = x0 + 1;

    float vy0 = (y0 >= 0 && y0 < HH) ? 1.f : 0.f;
    float vy1 = (y1 >= 0 && y1 < HH) ? 1.f : 0.f;
    float vx0 = (x0 >= 0 && x0 < WW) ? 1.f : 0.f;
    float vx1 = (x1 >= 0 && x1 < WW) ? 1.f : 0.f;

    float w00 = (1.f - ly) * (1.f - lx) * vy0 * vx0 * mk;
    float w01 = (1.f - ly) * lx         * vy0 * vx1 * mk;
    float w10 = ly         * (1.f - lx) * vy1 * vx0 * mk;
    float w11 = ly         * lx         * vy1 * vx1 * mk;

    int y0c = min(max(y0, 0), HH - 1), y1c = min(max(y1, 0), HH - 1);
    int x0c = min(max(x0, 0), WW - 1), x1c = min(max(x1, 0), WW - 1);
    int i00 = y0c * WW + x0c, i01 = y0c * WW + x1c;
    int i10 = y1c * WW + x0c, i11 = y1c * WW + x1c;

    const float* xb = x + (size_t)b * CC * HW;
    _Float16* dst = colsT + ((size_t)b * NN + n) * KDIM + kk * CC;

    #pragma unroll 4
    for (int c = 0; c < CC; ++c) {
        const float* xc = xb + c * HW;
        float v = w00 * xc[i00] + w01 * xc[i01] + w10 * xc[i10] + w11 * xc[i11];
        dst[c] = (_Float16)v;
    }
}

// ---------------------- kernel 3: weight -> f16 (reordered) ----------------
// Wf16[o][kk*256 + c] = weight[o][c][kk]
__global__ void dcn_wcvt(const float* __restrict__ weight,
                         _Float16* __restrict__ Wf16) {
    int idx = blockIdx.x * blockDim.x + threadIdx.x;
    if (idx >= OO * KDIM) return;
    int o = idx / KDIM;
    int t = idx % KDIM;
    int kk = t >> 8;          // t = kk*256 + c
    int c  = t & 255;
    Wf16[idx] = (_Float16)weight[(size_t)o * KDIM + c * 9 + kk];
}

// --------------------------- kernel 4: WMMA GEMM ---------------------------
// out[b][o][n] = sum_k Wf16[o][k] * colsT[b][n][k] + bias[o]
// block: 256 threads = 8 waves, tile 128(M) x 128(N), wave tile 32x64
// (2x4 accumulators, 8 WMMAs/step), K-step 32, v_wmma_f32_16x16x32_f16,
// TDM double-buffered LDS staging, loop unrolled x2 for in-place accum.
#define LDS_STRIDE 40        // f16 per tile row (80B: 16B-aligned, conflict-free)
#define KSTEPS     (KDIM / 32)   // 72 (even)

#if HAVE_TDM
// LDS byte address of a shared-memory pointer (generic -> AS(3) -> offset)
static __device__ inline unsigned lds_addr_of(const void* p) {
    return (unsigned)(unsigned long long)(const __attribute__((address_space(3))) char*)p;
}

// Issue a 2D TDM load: tileRows rows of 32 f16 (64B), row stride = strideElems
// f16 in global memory; LDS gets 16B pad per 64B row -> LDS_STRIDE=40 layout.
static __device__ inline void tdm_load_tile(unsigned ldsByteAddr, const void* gsrc,
                                            unsigned tileRows, unsigned strideElems) {
    unsigned long long ga = (unsigned long long)gsrc;
    u32x4 g0;
    g0[0] = 1u;                                            // count=1, user mode
    g0[1] = ldsByteAddr;                                   // lds_addr
    g0[2] = (unsigned)(ga & 0xFFFFFFFFu);                  // global_addr[31:0]
    g0[3] = (unsigned)((ga >> 32) & 0x1FFFFFFu)            // global_addr[56:32]
          | (2u << 30);                                    // type = 2 (image)
    const unsigned td0 = 1u << 20, td1 = 1u << 20;         // tensor dims (no OOB)
    i32x8 g1;
    g1[0] = (int)((1u << 16)        // data_size = 2 bytes
                | (1u << 20)        // pad_enable
                | (3u << 22)        // pad_interval: 16 DWORDs (one 64B row)
                | (3u << 25));      // pad_amount:   4 DWORDs (16B)
    g1[1] = (int)((td0 & 0xFFFFu) << 16);                          // tensor_dim0 lo
    g1[2] = (int)(((td0 >> 16) & 0xFFFFu) | ((td1 & 0xFFFFu) << 16));
    g1[3] = (int)(((td1 >> 16) & 0xFFFFu) | (32u << 16));          // tile_dim0 = 32
    g1[4] = (int)(tileRows & 0xFFFFu);                             // tile_dim1
    g1[5] = (int)strideElems;                                      // dim0_stride lo
    g1[6] = 0;                                                     // stride hi, dim1_stride
    g1[7] = 0;
    i32x4 z4 = {0, 0, 0, 0};
#if __clang_major__ >= 23
    i32x8 z8 = {0, 0, 0, 0, 0, 0, 0, 0};
    __builtin_amdgcn_tensor_load_to_lds(g0, g1, z4, z4, z8, 0);
#else
    __builtin_amdgcn_tensor_load_to_lds(g0, g1, z4, z4, 0);
#endif
}
#endif // HAVE_TDM

// one K-step of the wave tile: 2 A-frags, 4 B-frags, 8 WMMAs (in-place accum)
static __device__ __forceinline__ void gemm_step(const _Float16* __restrict__ lA,
                                                 const _Float16* __restrict__ lB,
                                                 int wm, int wn, int l16, int aoff,
                                                 v8f acc[2][4]) {
    FragAB fa[2], fb[4];
    #pragma unroll
    for (int i = 0; i < 2; ++i) {
        int base = (wm * 32 + i * 16 + l16) * LDS_STRIDE + aoff;
        fa[i].h[0] = *(const v8h*)&lA[base];
        fa[i].h[1] = *(const v8h*)&lA[base + 16];
    }
    #pragma unroll
    for (int j = 0; j < 4; ++j) {
        int base = (wn * 64 + j * 16 + l16) * LDS_STRIDE + aoff;
        fb[j].h[0] = *(const v8h*)&lB[base];
        fb[j].h[1] = *(const v8h*)&lB[base + 16];
    }
    #pragma unroll
    for (int i = 0; i < 2; ++i)
        #pragma unroll
        for (int j = 0; j < 4; ++j)
            acc[i][j] = __builtin_amdgcn_wmma_f32_16x16x32_f16(
                false, fa[i].v, false, fb[j].v,
                (short)0, acc[i][j], false, false);
}

__global__ void __launch_bounds__(256)
dcn_wmma_gemm(const _Float16* __restrict__ Wf16,
              const _Float16* __restrict__ colsT,
              const float* __restrict__ bias,
              float* __restrict__ out) {
    __shared__ __align__(16) _Float16 lsA[2][128 * LDS_STRIDE];   // 2 x 10 KB
    __shared__ __align__(16) _Float16 lsB[2][128 * LDS_STRIDE];   // 2 x 10 KB

    const int tid  = threadIdx.x;
    const int lane = tid & 31;
    const int wid  = tid >> 5;            // 0..7
    const int wm   = wid & 3;             // 4 waves along M (32 rows each)
    const int wn   = wid >> 2;            // 2 waves along N (64 cols each)
    const int l16  = lane & 15;
    const bool hi  = lane >= 16;
    const int aoff = hi ? 8 : 0;          // K 8..15 half for lanes 16..31

    const int n0 = blockIdx.x * 128;      // 32 n-tiles
    const int m0 = blockIdx.y * 128;      // 2 m-tiles
    const int b  = blockIdx.z;            // 4 batches

    const _Float16* Arow = Wf16 + (size_t)m0 * KDIM;
    const _Float16* Brow = colsT + ((size_t)b * NN + n0) * KDIM;

    v8f acc[2][4] = {};

#if HAVE_TDM
    if (wid == 0) {                       // prologue: tile 0 -> buffer 0
        tdm_load_tile(lds_addr_of(&lsA[0][0]), Arow, 128, KDIM);
        tdm_load_tile(lds_addr_of(&lsB[0][0]), Brow, 128, KDIM);
    }
#endif

    for (int kt = 0; kt < KSTEPS; kt += 2) {
        // ================= phase 0: compute buf0, prefetch -> buf1 =========
#if HAVE_TDM
        if (wid == 0) {                   // kt+1 < KSTEPS always (KSTEPS even)
            tdm_load_tile(lds_addr_of(&lsA[1][0]), Arow + (kt + 1) * 32, 128, KDIM);
            tdm_load_tile(lds_addr_of(&lsB[1][0]), Brow + (kt + 1) * 32, 128, KDIM);
            __builtin_amdgcn_s_wait_tensorcnt(2);   // tile kt's 2 DMAs done
        }
        __syncthreads();
#else
        #pragma unroll
        for (int s = tid; s < 512; s += 256) {
            int r = s >> 2, q = s & 3;
            *(v8h*)&lsA[0][r * LDS_STRIDE + q * 8] =
                *(const v8h*)(Arow + (size_t)r * KDIM + kt * 32 + q * 8);
            *(v8h*)&lsB[0][r * LDS_STRIDE + q * 8] =
                *(const v8h*)(Brow + (size_t)r * KDIM + kt * 32 + q * 8);
        }
        __syncthreads();
#endif
        gemm_step(&lsA[0][0], &lsB[0][0], wm, wn, l16, aoff, acc);
        __syncthreads();                  // reads of buf0 done

        // ================= phase 1: compute buf1, prefetch -> buf0 =========
#if HAVE_TDM
        if (wid == 0) {
            if (kt + 2 < KSTEPS) {
                tdm_load_tile(lds_addr_of(&lsA[0][0]), Arow + (kt + 2) * 32, 128, KDIM);
                tdm_load_tile(lds_addr_of(&lsB[0][0]), Brow + (kt + 2) * 32, 128, KDIM);
                __builtin_amdgcn_s_wait_tensorcnt(2);   // tile kt+1 done
            } else {
                __builtin_amdgcn_s_wait_tensorcnt(0);
            }
        }
        __syncthreads();
#else
        #pragma unroll
        for (int s = tid; s < 512; s += 256) {
            int r = s >> 2, q = s & 3;
            *(v8h*)&lsA[1][r * LDS_STRIDE + q * 8] =
                *(const v8h*)(Arow + (size_t)r * KDIM + (kt + 1) * 32 + q * 8);
            *(v8h*)&lsB[1][r * LDS_STRIDE + q * 8] =
                *(const v8h*)(Brow + (size_t)r * KDIM + (kt + 1) * 32 + q * 8);
        }
        __syncthreads();
#endif
        gemm_step(&lsA[1][0], &lsB[1][0], wm, wn, l16, aoff, acc);
        __syncthreads();                  // reads of buf1 done
    }

    // ---- epilogue: C/D layout -> lane = N col, VGPR r = M row (+8 if hi) ----
    #pragma unroll
    for (int i = 0; i < 2; ++i) {
        #pragma unroll
        for (int j = 0; j < 4; ++j) {
            int ncol = n0 + wn * 64 + j * 16 + l16;
            #pragma unroll
            for (int r = 0; r < 8; ++r) {
                int m = m0 + wm * 32 + i * 16 + r + (hi ? 8 : 0);
                out[(((size_t)b * OO + m) * NN) + ncol] = acc[i][j][r] + bias[m];
            }
        }
    }
}

// ------------------------------- launcher ----------------------------------
extern "C" void kernel_launch(void* const* d_in, const int* in_sizes, int n_in,
                              void* d_out, int out_size, void* d_ws, size_t ws_size,
                              hipStream_t stream) {
    const float* x      = (const float*)d_in[0];   // [4,256,64,64]
    const float* weight = (const float*)d_in[1];   // [256,256,3,3]
    const float* bias   = (const float*)d_in[2];   // [256]
    const float* w_om   = (const float*)d_in[3];   // [27,256,3,3]
    const float* b_om   = (const float*)d_in[4];   // [27]
    float* out = (float*)d_out;                    // [4,256,64,64]

    // workspace carve-up (all offsets 256B-aligned)
    char* ws = (char*)d_ws;
    float*    om    = (float*)ws;                               // 4*27*4096*4   = 1,769,472 B
    _Float16* Wf16  = (_Float16*)(ws + 1769472);                // 256*2304*2    = 1,179,648 B
    _Float16* colsT = (_Float16*)(ws + 1769472 + 1179648);      // 4*4096*2304*2 = 75,497,472 B

    {   // k1: offset/mask conv
        int total = BB * 27 * NN;
        dcn_om_conv<<<(total + 255) / 256, 256, 0, stream>>>(x, w_om, b_om, om);
    }
    {   // k3: weight convert (independent of k1/k2)
        int total = OO * KDIM;
        dcn_wcvt<<<(total + 255) / 256, 256, 0, stream>>>(weight, Wf16);
    }
    {   // k2: deformable sampling -> colsT (f16)
        int total = BB * KK9 * NN;
        dcn_im2col<<<(total + 255) / 256, 256, 0, stream>>>(x, om, colsT);
    }
    {   // k4: WMMA GEMM + bias, TDM double-buffered
        dim3 grid(NN / 128, OO / 128, BB);   // (32, 2, 4)
        dcn_wmma_gemm<<<grid, 256, 0, stream>>>(Wf16, colsT, bias, out);
    }
}
